// Net_1683627180173
// MI455X (gfx1250) — matrix operand order repaired and checked by reading, not compiled, not run on hardware
//
#include <hip/hip_runtime.h>
#include <hip/hip_bf16.h>
#include <math.h>

typedef __attribute__((ext_vector_type(16))) _Float16 v16h;
typedef __attribute__((ext_vector_type(8)))  float    v8f;

#define F_IN 128
#define HID  16
#define NCLS 10

// ---------------- degree / normalization ----------------
__global__ void deg_init_kernel(float* __restrict__ deg, int N) {
  int i = blockIdx.x * blockDim.x + threadIdx.x;
  if (i < N) deg[i] = 1.0f;  // self-loop contributes 1
}

__global__ void deg_count_kernel(const int* __restrict__ col, float* __restrict__ deg, int E) {
  int i = blockIdx.x * blockDim.x + threadIdx.x;
  if (i < E) atomicAdd(&deg[col[i]], 1.0f);
}

__global__ void deg_rsqrt_kernel(float* __restrict__ deg, int N) {
  int i = blockIdx.x * blockDim.x + threadIdx.x;
  if (i < N) deg[i] = rsqrtf(deg[i]);   // deg -> dis = deg^{-1/2}
}

__global__ void zero_kernel(float* __restrict__ p, long long n) {
  long long i = (long long)blockIdx.x * blockDim.x + threadIdx.x;
  long long stride = (long long)gridDim.x * blockDim.x;
  for (; i < n; i += stride) p[i] = 0.0f;
}

// ---------------- GEMM1: H1 = X(100k x 128) @ W1(128 x 16), f16 WMMA ----------------
// One wave computes a 16x16 output tile via 4x v_wmma_f32_16x16x32_f16 over K=128.
// W1 is pre-packed in LDS in B-fragment order: each lane reads its full 32-byte
// fragment with one v16h load (2x ds_load_b128) per K-chunk.
__global__ __launch_bounds__(256) void gemm1_wmma_kernel(
    const float* __restrict__ X, const float* __restrict__ W1,
    float* __restrict__ H1, int N) {
  // w1p[c][l][h] = W1[(c*32 + (h>>3)*16 + (l>>4)*8 + (h&7))*16 + (l&15)]  (c=chunk, l=lane, h=half)
  __shared__ __align__(32) _Float16 w1p[4 * 32 * 16];
  int t = threadIdx.x;
  for (int i = t; i < 4 * 32 * 16; i += 256) {
    int c = i >> 9;
    int l = (i >> 4) & 31;
    int h = i & 15;
    int kb = c * 32 + ((h >> 3) << 4) + ((l >> 4) << 3) + (h & 7);
    w1p[i] = (_Float16)W1[kb * HID + (l & 15)];
  }
  __syncthreads();

  int wave = t >> 5;
  int lane = t & 31;
  int tile = blockIdx.x * 8 + wave;              // 16-row tile index
  int m  = lane & 15;                            // A row / D column within tile
  int hi = lane >> 4;                            // lane half select
  int row = tile * 16 + m;
  long long srow = (row < N) ? row : (long long)(N - 1);  // clamp (stores guarded)

  const v16h* bfrag = (const v16h*)w1p;
  v8f acc = {};
#pragma unroll
  for (int c = 0; c < 4; ++c) {
    int k0 = c * 32;
    v16h a;
#pragma unroll
    for (int j = 0; j < 8; ++j) {
      // A layout: lanes0-15 K {0-7,16-23}, lanes16-31 K {8-15,24-31}; 2 K per VGPR
      int ka = k0 + hi * 8 + ((j >> 2) << 4) + ((j & 3) << 1);
      float2 xa = *(const float2*)(X + srow * F_IN + ka);
      a[2 * j]     = (_Float16)xa.x;
      a[2 * j + 1] = (_Float16)xa.y;
    }
    v16h b = bfrag[c * 32 + lane];
    acc = __builtin_amdgcn_wmma_f32_16x16x32_f16(
        false, a, false, b, (short)0, acc, false, false);
  }
#pragma unroll
  for (int r = 0; r < 8; ++r) {                  // D: M = r + 8*hi, N = m
    int gr = tile * 16 + r + hi * 8;
    if (gr < N) H1[(long long)gr * HID + m] = acc[r];
  }
}

// ---------------- GEMM2: LP = H(100k x 16) @ W2(16 x 10), one padded WMMA ----------------
__global__ __launch_bounds__(256) void gemm2_wmma_kernel(
    const float* __restrict__ H, const float* __restrict__ W2,
    float* __restrict__ LP, int N) {
  // Pre-packed, zero-padded B fragment: w2p[l][h] = W2pad[((h>>3)*16 + (l>>4)*8 + (h&7))*16 + (l&15)]
  __shared__ __align__(32) _Float16 w2p[32 * 16];
  int t = threadIdx.x;
  for (int i = t; i < 32 * 16; i += 256) {
    int l = i >> 4;
    int h = i & 15;
    int n = l & 15;
    int kb = ((h >> 3) << 4) + ((l >> 4) << 3) + (h & 7);
    w2p[i] = (kb < HID && n < NCLS) ? (_Float16)W2[kb * NCLS + n] : (_Float16)0.0f;
  }
  __syncthreads();

  int wave = t >> 5;
  int lane = t & 31;
  int tile = blockIdx.x * 8 + wave;
  int m  = lane & 15;
  int hi = lane >> 4;
  int row = tile * 16 + m;
  long long srow = (row < N) ? row : (long long)(N - 1);

  v16h a;
#pragma unroll
  for (int j = 0; j < 8; ++j) {
    int ka = hi * 8 + ((j >> 2) << 4) + ((j & 3) << 1);
    float2 ha;
    if (ka < HID) {
      ha = *(const float2*)(H + srow * HID + ka);
    } else {
      ha.x = 0.0f; ha.y = 0.0f;                  // K padding 16..31
    }
    a[2 * j]     = (_Float16)ha.x;
    a[2 * j + 1] = (_Float16)ha.y;
  }
  v16h b = ((const v16h*)w2p)[lane];
  v8f acc = {};
  acc = __builtin_amdgcn_wmma_f32_16x16x32_f16(
      false, a, false, b, (short)0, acc, false, false);
#pragma unroll
  for (int r = 0; r < 8; ++r) {
    int gr = tile * 16 + r + hi * 8;
    if (gr < N) LP[(long long)gr * 16 + m] = acc[r];   // stride 16 (cols 10..15 = 0)
  }
}

// ---------------- edge scatter-add: dst[col] += src[row] * dis[row]*dis[col] ----------------
template <int F, int SSTR, int DSTR>
__global__ void scatter_kernel(const int* __restrict__ row, const int* __restrict__ col,
                               const float* __restrict__ dis, const float* __restrict__ src,
                               float* __restrict__ dst, int E) {
  long long tid = (long long)blockIdx.x * blockDim.x + threadIdx.x;
  long long total = (long long)E * F;
  long long stride = (long long)gridDim.x * blockDim.x;
  for (; tid < total; tid += stride) {
    int e = (int)(tid / F);
    int f = (int)(tid - (long long)e * F);
    int r = row[e], c = col[e];
    float v = src[(long long)r * SSTR + f] * dis[r] * dis[c];
    atomicAdd(&dst[(long long)c * DSTR + f], v);   // resolves in 192MB L2
  }
}

// ---------------- layer-1 finish: fold self-loop, bias, ReLU (in place on AGG) ----------------
__global__ void finish1_kernel(const float* __restrict__ H1, float* __restrict__ AGG,
                               const float* __restrict__ dis, const float* __restrict__ b1,
                               int N) {
  long long tid = (long long)blockIdx.x * blockDim.x + threadIdx.x;
  if (tid < (long long)N * HID) {
    int i = (int)(tid >> 4), f = (int)(tid & 15);
    float d = dis[i];
    float v = AGG[tid] + H1[tid] * d * d + b1[f];
    AGG[tid] = fmaxf(v, 0.0f);
  }
}

// ---------------- layer-2 finish: self-loop, bias, log_softmax over 10 classes ----------------
__global__ void finish2_kernel(const float* __restrict__ LP, const float* __restrict__ AGG2,
                               const float* __restrict__ dis, const float* __restrict__ b2,
                               float* __restrict__ OUT, int N) {
  int i = blockIdx.x * blockDim.x + threadIdx.x;
  if (i >= N) return;
  float d2 = dis[i] * dis[i];
  float v[NCLS];
  float mx = -1e30f;
#pragma unroll
  for (int c = 0; c < NCLS; ++c) {
    v[c] = AGG2[(long long)i * NCLS + c] + LP[(long long)i * 16 + c] * d2 + b2[c];
    mx = fmaxf(mx, v[c]);
  }
  float s = 0.0f;
#pragma unroll
  for (int c = 0; c < NCLS; ++c) s += expf(v[c] - mx);
  float lse = mx + logf(s);
#pragma unroll
  for (int c = 0; c < NCLS; ++c) OUT[(long long)i * NCLS + c] = v[c] - lse;
}

// ---------------- launch ----------------
extern "C" void kernel_launch(void* const* d_in, const int* in_sizes, int n_in,
                              void* d_out, int out_size, void* d_ws, size_t ws_size,
                              hipStream_t stream) {
  const float* X  = (const float*)d_in[0];
  const int*   EI = (const int*)d_in[1];
  const float* W1 = (const float*)d_in[2];
  const float* B1 = (const float*)d_in[3];
  const float* W2 = (const float*)d_in[4];
  const float* B2 = (const float*)d_in[5];
  float* OUT = (float*)d_out;

  int N = in_sizes[0] / F_IN;
  int E = in_sizes[1] / 2;
  const int* row = EI;          // edge_index[0] = sources
  const int* col = EI + E;      // edge_index[1] = targets

  // workspace layout (floats): dis[N] | H1[N*16] | AGG1[N*16] | LP[N*16] | AGG2[N*10]
  float* ws   = (float*)d_ws;
  float* dis  = ws;
  float* h1   = dis  + (long long)N;
  float* agg1 = h1   + (long long)N * HID;
  float* lp   = agg1 + (long long)N * HID;
  float* agg2 = lp   + (long long)N * 16;

  const int B = 256;

  // degrees -> dis = rsqrt(deg)
  deg_init_kernel<<<(N + B - 1) / B, B, 0, stream>>>(dis, N);
  deg_count_kernel<<<(E + B - 1) / B, B, 0, stream>>>(col, dis, E);
  deg_rsqrt_kernel<<<(N + B - 1) / B, B, 0, stream>>>(dis, N);

  // GEMM1 (WMMA): 8 tiles (16 rows each) per 256-thread block
  int tiles = (N + 15) / 16;
  int gblocks = (tiles + 7) / 8;
  gemm1_wmma_kernel<<<gblocks, 256, 0, stream>>>(X, W1, h1, N);

  // layer-1 aggregation
  zero_kernel<<<2048, B, 0, stream>>>(agg1, (long long)N * HID);
  {
    long long tot = (long long)E * HID;
    long long blk = (tot + B - 1) / B;
    int sblocks = (blk > 1048576LL) ? 1048576 : (int)blk;
    scatter_kernel<HID, HID, HID><<<sblocks, B, 0, stream>>>(row, col, dis, h1, agg1, E);
  }
  {
    long long tot = (long long)N * HID;
    finish1_kernel<<<(int)((tot + B - 1) / B), B, 0, stream>>>(h1, agg1, dis, B1, N);
  }

  // GEMM2 (WMMA, K/N zero-padded)
  gemm2_wmma_kernel<<<gblocks, 256, 0, stream>>>(agg1, W2, lp, N);

  // layer-2 aggregation + log_softmax
  zero_kernel<<<2048, B, 0, stream>>>(agg2, (long long)N * NCLS);
  {
    long long tot = (long long)E * NCLS;
    long long blk = (tot + B - 1) / B;
    int sblocks = (blk > 1048576LL) ? 1048576 : (int)blk;
    scatter_kernel<NCLS, 16, NCLS><<<sblocks, B, 0, stream>>>(row, col, dis, lp, agg2, E);
  }
  finish2_kernel<<<(N + B - 1) / B, B, 0, stream>>>(lp, agg2, dis, B2, OUT, N);
}
